// CrossAttention_63256278335564
// MI455X (gfx1250) — matrix operand order repaired and checked by reading, not compiled, not run on hardware
//
#include <hip/hip_runtime.h>

typedef __bf16 bf16;
typedef __attribute__((ext_vector_type(16))) __bf16 v16bf;
typedef __attribute__((ext_vector_type(8)))  __bf16 v8bf;
typedef __attribute__((ext_vector_type(8)))  float  v8f;

static constexpr int BATCH = 4;
static constexpr int S1 = 1024;
static constexpr int S2 = 1024;
static constexpr int IND = 1024;   // IN1 == IN2 == H*D
static constexpr int H  = 16;
static constexpr int D  = 64;

#define HAS_TDM (__has_builtin(__builtin_amdgcn_tensor_load_to_lds) && \
                 __has_builtin(__builtin_amdgcn_s_wait_tensorcnt))

// TDM writes LDS behind the compiler's back (descriptor-based DMA). Publish the
// LDS object to a memory-clobbering asm so it is considered escaped+written;
// otherwise loads from "never-stored" LDS fold to poison and vanish.
__device__ __forceinline__ void lds_publish(const void* p) {
  asm volatile("" : : "s"(p) : "memory");
}

#if HAS_TDM
typedef __attribute__((ext_vector_type(4))) unsigned int v4u;
typedef __attribute__((ext_vector_type(8))) int          v8i;
typedef __attribute__((ext_vector_type(4))) int          v4i;

// Issue one TDM 2-D tile load: global (strided rows) -> LDS (with optional row padding).
// D# per CDNA5 ISA ch.8: group0 = {count, lds_addr, global_addr, type=2},
// group1 = {data_size=2B, pad cfg, tensor dims, tile dims, dim0 stride}.
__device__ __forceinline__ void tdm_load_2d(
    unsigned lds_off, const void* gptr,
    unsigned tile_w, unsigned tile_h,        // tile size in bf16 elems / rows
    unsigned tensor_w, unsigned tensor_h,    // tensor dims (elems) for OOB
    unsigned stride_elems,                   // row stride in elems
    unsigned pad_interval_code, unsigned pad_amount_code, unsigned pad_en)
{
  unsigned long long ga = (unsigned long long)(size_t)gptr;
  v4u g0;
  g0.x = 1u;                                             // count=1, user mode
  g0.y = lds_off;                                        // lds_addr
  g0.z = (unsigned)(ga & 0xFFFFFFFFu);                   // global_addr[31:0]
  g0.w = (unsigned)((ga >> 32) & 0x1FFFFFFu) | (2u << 30); // addr[56:32] | type=2
  v8i g1;
  g1[0] = (int)((1u << 16) | (pad_en << 20) |
                (pad_interval_code << 22) | (pad_amount_code << 25)); // data_size=1(2B)
  g1[1] = (int)((tensor_w & 0xFFFFu) << 16);                          // dim0 lo16
  g1[2] = (int)(((tensor_w >> 16) & 0xFFFFu) | ((tensor_h & 0xFFFFu) << 16));
  g1[3] = (int)(((tensor_h >> 16) & 0xFFFFu) | ((tile_w & 0xFFFFu) << 16));
  g1[4] = (int)(tile_h & 0xFFFFu);                                    // tile_dim1, tile_dim2=0
  g1[5] = (int)stride_elems;                                          // dim0_stride lo32
  g1[6] = 0;
  g1[7] = 0;
  v4i gz4 = {0, 0, 0, 0};
  v8i gz8 = {0, 0, 0, 0, 0, 0, 0, 0};
  __builtin_amdgcn_tensor_load_to_lds(g0, g1, gz4, gz4, gz8, 0);
}
#endif

// ---------------- WMMA helpers (CDNA5 wave32 layouts) ----------------
__device__ __forceinline__ v16bf load_afrag(const bf16* p, int ksel) {
  v8bf lo = *(const v8bf*)(p + ksel * 8);
  v8bf hi = *(const v8bf*)(p + ksel * 8 + 16);
  v16bf a;
#pragma unroll
  for (int i = 0; i < 8; ++i) { a[i] = lo[i]; a[i + 8] = hi[i]; }
  return a;
}
__device__ __forceinline__ v16bf load_bfrag(const bf16* p, int ksel) {
  v8bf p0 = *(const v8bf*)(p + ksel * 8);
  v8bf p1 = *(const v8bf*)(p + ksel * 8 + 16);
  v16bf b;
#pragma unroll
  for (int i = 0; i < 8; ++i) { b[i] = p0[i]; b[i + 8] = p1[i]; }
  return b;
}
__device__ __forceinline__ v8f wmma_bf16(v16bf a, v16bf b, v8f c) {
  return __builtin_amdgcn_wmma_f32_16x16x32_bf16(false, a, false, b, (short)0, c, false, false);
}

// ---------------- fp32 -> bf16 conversion ----------------
__global__ __launch_bounds__(256) void cvt_f32_bf16(const float* __restrict__ src,
                                                    bf16* __restrict__ dst, int n) {
  int i = blockIdx.x * 256 + threadIdx.x;
  if (i < n) dst[i] = (bf16)src[i];
}

__global__ __launch_bounds__(256) void zero_f32(float* __restrict__ p, int n) {
  int i = blockIdx.x * 256 + threadIdx.x;
  if (i < n) p[i] = 0.0f;
}

// ---------------- GEMM: C[M,N] = X[M,K] * W[N,K]^T (bf16 in, f32 accum) ----------------
// 128x128 tile per 256-thread block (8 wave32s, 2x4 frags each), K step 32.
// Tile staging via Tensor Data Mover when available.
__global__ __launch_bounds__(256) void gemm_xwT_bf16(
    const bf16* __restrict__ X, const bf16* __restrict__ W,
    void* __restrict__ Cout, int M, int N, int Kd, int mode,
    const float* __restrict__ bias, const float* __restrict__ xcls,
    int xcls_stride, int rows_per_batch)
{
  __shared__ bf16 As[128][40];   // 80B row stride: 64B data + 16B pad
  __shared__ bf16 Bs[128][40];
  constexpr unsigned LDS_A_OFF = 0;
  constexpr unsigned LDS_B_OFF = 128 * 40 * 2;   // assumes As,Bs packed in decl order

  const int tid  = threadIdx.x;
  const int wave = tid >> 5, lane = tid & 31;
  const int lh   = lane & 15, ksel = lane >> 4;
  const int wm   = wave >> 1, wn = wave & 1;
  const int m0   = blockIdx.x * 128, n0 = blockIdx.y * 128;

  const v8f vz = {0.f, 0.f, 0.f, 0.f, 0.f, 0.f, 0.f, 0.f};
  v8f acc[2][4];
#pragma unroll
  for (int mi = 0; mi < 2; ++mi)
#pragma unroll
    for (int ni = 0; ni < 4; ++ni) acc[mi][ni] = vz;

  for (int kb = 0; kb < Kd; kb += 32) {
    __syncthreads();
#if HAS_TDM
    if (tid < 32) {   // wave 0 drives the DMA engine; EXEC is ignored by TDM
      // pad codes: interval 3 -> 16 DWORDs (64B), amount 3 -> 4 DWORDs (16B)
      tdm_load_2d(LDS_A_OFF, X + (size_t)m0 * Kd + kb, 32, 128,
                  (unsigned)Kd, (unsigned)M, (unsigned)Kd, 3, 3, 1);
      tdm_load_2d(LDS_B_OFF, W + (size_t)n0 * Kd + kb, 32, 128,
                  (unsigned)Kd, (unsigned)N, (unsigned)Kd, 3, 3, 1);
      __builtin_amdgcn_s_wait_tensorcnt(0);
    }
    lds_publish(&As[0][0]);
    lds_publish(&Bs[0][0]);
#else
#pragma unroll
    for (int p = 0; p < 2; ++p) {
      int idx = (tid + p * 256) * 8;
      int r = idx >> 5, c = idx & 31;
      const bf16* xa = X + (size_t)(m0 + r) * Kd + kb + c;
      const bf16* wb = W + (size_t)(n0 + r) * Kd + kb + c;
      *(v8bf*)&As[r][c] = *(const v8bf*)xa;
      *(v8bf*)&Bs[r][c] = *(const v8bf*)wb;
      __builtin_prefetch(xa + 32, 0, 1);   // speculative: OOB is dropped silently
      __builtin_prefetch(wb + 32, 0, 1);
    }
#endif
    __syncthreads();
#pragma unroll
    for (int mi = 0; mi < 2; ++mi) {
      v16bf a = load_afrag(&As[wm * 32 + mi * 16 + lh][0], ksel);
#pragma unroll
      for (int ni = 0; ni < 4; ++ni) {
        v16bf b = load_bfrag(&Bs[wn * 64 + ni * 16 + lh][0], ksel);
        acc[mi][ni] = wmma_bf16(a, b, acc[mi][ni]);
      }
    }
  }

  const int rofs = ksel << 3;
#pragma unroll
  for (int mi = 0; mi < 2; ++mi)
#pragma unroll
    for (int ni = 0; ni < 4; ++ni)
#pragma unroll
      for (int v = 0; v < 8; ++v) {
        int row = m0 + wm * 32 + mi * 16 + v + rofs;
        int col = n0 + wn * 64 + ni * 16 + lh;
        float val = acc[mi][ni][v];
        if (mode == 0) {
          ((bf16*)Cout)[(size_t)row * N + col] = (bf16)val;
        } else {
          int bidx = row / rows_per_batch;
          ((float*)Cout)[(size_t)row * N + col] =
              val + bias[col] + xcls[(size_t)bidx * xcls_stride + col];
        }
      }
}

// ---------------- fused attention: one block = (b, h, 64 rows of S1) ----------------
// Full-row score buffer (64 x 1024 f32 = 256KB) lives in CDNA5's 320KB LDS -> plain softmax.
__global__ __launch_bounds__(256) void attn_kernel(
    const bf16* __restrict__ Q, const bf16* __restrict__ Kmat,
    const bf16* __restrict__ V, const int* __restrict__ mask,
    bf16* __restrict__ O)
{
  constexpr int SPAD = 1032;                 // padded f32 row stride
  extern __shared__ char smem[];
  float* Ssc = (float*)smem;                           // [64][SPAD] f32
  bf16* qs   = (bf16*)(smem + (size_t)64 * SPAD * 4);  // [64][72] bf16
  bf16* kv   = qs + 64 * 72;                           // [64][72] bf16 (K then V staging)
  constexpr unsigned QS_OFF = 64u * SPAD * 4u;         // dynamic LDS base is offset 0
  constexpr unsigned KV_OFF = QS_OFF + 64u * 72u * 2u;

  const int tid  = threadIdx.x;
  const int wave = tid >> 5, lane = tid & 31;
  const int lh   = lane & 15, ksel = lane >> 4;
  const int rofs = ksel << 3;
  const int b    = blockIdx.x >> 4;
  const int h    = blockIdx.x & 15;
  const int s0   = blockIdx.y * 64;
  const int mrow = (wave >> 1) * 16;   // 16-row strip of the 64-row tile
  const int ncb  = (wave & 1) * 32;    // 32-col strip

  const v8f vz = {0.f, 0.f, 0.f, 0.f, 0.f, 0.f, 0.f, 0.f};

  // stage q tile (64 x 64): pad codes: interval 4 -> 32 DWORDs (128B), amount 3 -> 16B
#if HAS_TDM
  if (tid < 32) {
    tdm_load_2d(QS_OFF, Q + ((size_t)(b * S1 + s0)) * IND + h * D, 64, 64,
                IND, BATCH * S1, IND, 4, 3, 1);
    __builtin_amdgcn_s_wait_tensorcnt(0);
  }
  lds_publish(qs);
#else
#pragma unroll
  for (int p = 0; p < 2; ++p) {
    int idx = (tid + p * 256) * 8;
    int r = idx >> 6, c = idx & 63;
    *(v8bf*)(qs + r * 72 + c) =
        *(const v8bf*)(Q + ((size_t)(b * S1 + s0 + r)) * IND + h * D + c);
  }
#endif

  // ---- phase 1: S = (q . k^T) / sqrt(D) ----
  for (int t0 = 0; t0 < S2; t0 += 64) {
    __syncthreads();
#if HAS_TDM
    if (tid < 32) {
      tdm_load_2d(KV_OFF, Kmat + ((size_t)(b * S2 + t0)) * IND + h * D, 64, 64,
                  IND, BATCH * S2, IND, 4, 3, 1);
      __builtin_amdgcn_s_wait_tensorcnt(0);
    }
    lds_publish(kv);
#else
#pragma unroll
    for (int p = 0; p < 2; ++p) {
      int idx = (tid + p * 256) * 8;
      int r = idx >> 6, c = idx & 63;
      *(v8bf*)(kv + r * 72 + c) =
          *(const v8bf*)(Kmat + ((size_t)(b * S2 + t0 + r)) * IND + h * D + c);
    }
#endif
    __syncthreads();
    v8f sacc[2] = {vz, vz};
#pragma unroll
    for (int d0 = 0; d0 < 64; d0 += 32) {
      v16bf a = load_afrag(qs + (mrow + lh) * 72 + d0, ksel);
#pragma unroll
      for (int ni = 0; ni < 2; ++ni) {
        v16bf bb = load_bfrag(kv + (ncb + ni * 16 + lh) * 72 + d0, ksel);
        sacc[ni] = wmma_bf16(a, bb, sacc[ni]);
      }
    }
#pragma unroll
    for (int ni = 0; ni < 2; ++ni)
#pragma unroll
      for (int v = 0; v < 8; ++v)
        Ssc[(size_t)(mrow + v + rofs) * SPAD + t0 + ncb + ni * 16 + lh] =
            sacc[ni][v] * 0.125f;
  }
  __syncthreads();

  // ---- phase 2: masked softmax, each wave owns 8 full rows ----
  const int* mrow_mask = mask + b * S2;
  for (int rr = wave * 8; rr < wave * 8 + 8; ++rr) {
    float* srow = Ssc + (size_t)rr * SPAD;
    float mx = -3.0e38f;
    for (int j = lane; j < S2; j += 32) {
      float s = srow[j];
      if (mrow_mask[j] == 0) s = -1.0e9f;
      srow[j] = s;
      mx = fmaxf(mx, s);
    }
#pragma unroll
    for (int off = 16; off > 0; off >>= 1) mx = fmaxf(mx, __shfl_xor(mx, off, 32));
    float sum = 0.f;
    for (int j = lane; j < S2; j += 32) {
      float e = __expf(srow[j] - mx);
      srow[j] = e;
      sum += e;
    }
#pragma unroll
    for (int off = 16; off > 0; off >>= 1) sum += __shfl_xor(sum, off, 32);
    float inv = 1.0f / sum;
    for (int j = lane; j < S2; j += 32) srow[j] *= inv;
  }
  __syncthreads();

  // ---- phase 3: O = P . V ----
  v8f oacc[2] = {vz, vz};
  for (int t0 = 0; t0 < S2; t0 += 64) {
    __syncthreads();
#if HAS_TDM
    if (tid < 32) {
      tdm_load_2d(KV_OFF, V + ((size_t)(b * S2 + t0)) * IND + h * D, 64, 64,
                  IND, BATCH * S2, IND, 4, 3, 1);
      __builtin_amdgcn_s_wait_tensorcnt(0);
    }
    lds_publish(kv);
#else
#pragma unroll
    for (int p = 0; p < 2; ++p) {
      int idx = (tid + p * 256) * 8;
      int r = idx >> 6, c = idx & 63;
      *(v8bf*)(kv + r * 72 + c) =
          *(const v8bf*)(V + ((size_t)(b * S2 + t0 + r)) * IND + h * D + c);
    }
#endif
    __syncthreads();
#pragma unroll
    for (int ts = 0; ts < 64; ts += 32) {
      v16bf a;
      const float* sp = Ssc + (size_t)(mrow + lh) * SPAD + t0 + ts + ksel * 8;
#pragma unroll
      for (int i = 0; i < 8; ++i) { a[i] = (bf16)sp[i]; a[i + 8] = (bf16)sp[i + 16]; }
#pragma unroll
      for (int ni = 0; ni < 2; ++ni) {
        v16bf bb;
        const bf16* vp = kv + (size_t)(ts + ksel * 8) * 72 + ncb + ni * 16 + lh;
#pragma unroll
        for (int i = 0; i < 16; ++i) bb[i] = vp[(size_t)(i + (i & 8)) * 72];
        oacc[ni] = wmma_bf16(a, bb, oacc[ni]);
      }
    }
  }
#pragma unroll
  for (int ni = 0; ni < 2; ++ni)
#pragma unroll
    for (int v = 0; v < 8; ++v) {
      int s = s0 + mrow + v + rofs;
      int d = ncb + ni * 16 + lh;
      O[((size_t)(b * S1 + s)) * IND + h * D + d] = (bf16)oacc[ni][v];
    }
}

// ---------------- LayerNorm per row + mean over S1 via atomics ----------------
__global__ __launch_bounds__(256) void ln_mean_kernel(
    const float* __restrict__ Y, const float* __restrict__ gamma,
    const float* __restrict__ beta, float* __restrict__ out)
{
  __shared__ float rs[8], rss[8], tot[2];
  const int row = blockIdx.x;
  const int b   = row >> 10;
  const float* y = Y + (size_t)row * IND;
  const int tid = threadIdx.x, wave = tid >> 5, lane = tid & 31;

  float v[4], s = 0.f, ss = 0.f;
#pragma unroll
  for (int k = 0; k < 4; ++k) {
    int c = tid + k * 256;
    float x = y[c];
    v[k] = x; s += x; ss += x * x;
  }
#pragma unroll
  for (int off = 16; off > 0; off >>= 1) {
    s  += __shfl_xor(s, off, 32);
    ss += __shfl_xor(ss, off, 32);
  }
  if (lane == 0) { rs[wave] = s; rss[wave] = ss; }
  __syncthreads();
  if (tid == 0) {
    float a = 0.f, q = 0.f;
    for (int i = 0; i < 8; ++i) { a += rs[i]; q += rss[i]; }
    tot[0] = a; tot[1] = q;
  }
  __syncthreads();
  const float mu  = tot[0] * (1.0f / IND);
  const float var = tot[1] * (1.0f / IND) - mu * mu;
  const float inv = rsqrtf(var + 1e-5f);
#pragma unroll
  for (int k = 0; k < 4; ++k) {
    int c = tid + k * 256;
    float val = (v[k] - mu) * inv * gamma[c] + beta[c];
    atomicAdd(out + (size_t)b * IND + c, val * (1.0f / S1));
  }
}

// ---------------- host launcher ----------------
extern "C" void kernel_launch(void* const* d_in, const int* in_sizes, int n_in,
                              void* d_out, int out_size, void* d_ws, size_t ws_size,
                              hipStream_t stream) {
  const float* x1    = (const float*)d_in[0];
  const float* x2    = (const float*)d_in[1];
  const int*   mask  = (const int*)  d_in[2];
  const float* Wq    = (const float*)d_in[3];
  const float* Wk    = (const float*)d_in[4];
  const float* Wv    = (const float*)d_in[5];
  const float* Wo    = (const float*)d_in[6];
  const float* bo    = (const float*)d_in[7];
  const float* gamma = (const float*)d_in[8];
  const float* beta  = (const float*)d_in[9];
  float* out = (float*)d_out;

  char* ws = (char*)d_ws;
  constexpr size_t NX  = (size_t)BATCH * S1 * IND;
  constexpr size_t NW  = (size_t)IND * IND;
  bf16* x1b = (bf16*)(ws);
  bf16* x2b = (bf16*)(ws + 8388608);
  bf16* Wqb = (bf16*)(ws + 16777216);
  bf16* Wkb = (bf16*)(ws + 18874368);
  bf16* Wvb = (bf16*)(ws + 20971520);
  bf16* Wob = (bf16*)(ws + 23068672);
  bf16* Qb  = (bf16*)(ws + 25165824);
  bf16* Kb  = (bf16*)(ws + 33554432);
  bf16* Vb  = (bf16*)(ws + 41943040);
  bf16* Ab  = (bf16*)(ws + 50331648);
  float* Yf = (float*)(ws + 58720256);

  zero_f32<<<(BATCH * IND + 255) / 256, 256, 0, stream>>>(out, BATCH * IND);

  cvt_f32_bf16<<<(int)(NX / 256), 256, 0, stream>>>(x1, x1b, (int)NX);
  cvt_f32_bf16<<<(int)(NX / 256), 256, 0, stream>>>(x2, x2b, (int)NX);
  cvt_f32_bf16<<<(int)(NW / 256), 256, 0, stream>>>(Wq, Wqb, (int)NW);
  cvt_f32_bf16<<<(int)(NW / 256), 256, 0, stream>>>(Wk, Wkb, (int)NW);
  cvt_f32_bf16<<<(int)(NW / 256), 256, 0, stream>>>(Wv, Wvb, (int)NW);
  cvt_f32_bf16<<<(int)(NW / 256), 256, 0, stream>>>(Wo, Wob, (int)NW);

  const int M = BATCH * S1;   // 4096
  dim3 ggrid(M / 128, IND / 128);
  gemm_xwT_bf16<<<ggrid, 256, 0, stream>>>(x1b, Wqb, Qb, M, IND, IND, 0, nullptr, nullptr, 0, S1);
  gemm_xwT_bf16<<<ggrid, 256, 0, stream>>>(x2b, Wkb, Kb, M, IND, IND, 0, nullptr, nullptr, 0, S2);
  gemm_xwT_bf16<<<ggrid, 256, 0, stream>>>(x2b, Wvb, Vb, M, IND, IND, 0, nullptr, nullptr, 0, S2);

  const size_t attn_lds = (size_t)64 * 1032 * 4 + 2 * (size_t)64 * 72 * 2;
  attn_kernel<<<dim3(BATCH * H, S1 / 64), 256, attn_lds, stream>>>(Qb, Kb, Vb, mask, Ab);

  gemm_xwT_bf16<<<ggrid, 256, 0, stream>>>(Ab, Wob, Yf, M, IND, IND, 1, bo, x2, S2 * IND, S1);

  ln_mean_kernel<<<M, 256, 0, stream>>>(Yf, gamma, beta, out);
}
